// RoleSensitiveEmbedding_28621662060563
// MI455X (gfx1250) — matrix operand order repaired and checked by reading, not compiled, not run on hardware
//
#include <hip/hip_runtime.h>
#include <hip/hip_bf16.h>
#include <stdint.h>

// ---- types -----------------------------------------------------------------
typedef __attribute__((ext_vector_type(16))) __bf16        v16bf;
typedef __attribute__((ext_vector_type(8)))  float         v8f;
typedef __attribute__((ext_vector_type(4)))  unsigned int  u32x4;
typedef __attribute__((ext_vector_type(2)))  unsigned int  u32x2;
typedef __attribute__((ext_vector_type(4)))  float         f32x4;

union FragU { u32x4 q[2]; v16bf v; };

#define DM         1024      // D_MODEL (K and N)
#define KSTEP      32
#define MROWS      32        // tokens per block (2 WMMA M-tiles)
#define LDS_STRIDE 1032      // 1024 + 8 halves pad: row stride 2064B = 516 dwords (516 % 64 == 4)

__device__ __forceinline__ unsigned int f2bf(float f) {
  unsigned int u = __float_as_uint(f);
  u += 0x7FFFu + ((u >> 16) & 1u);          // round-to-nearest-even
  return u >> 16;
}

// ---- W0/W1 fp32 -> bf16 pre-pass ------------------------------------------
__global__ __launch_bounds__(256) void cvt_w_kernel(
    const float* __restrict__ W0, const float* __restrict__ W1,
    unsigned short* __restrict__ Wb0, unsigned short* __restrict__ Wb1)
{
  const unsigned i = (blockIdx.x * 256u + threadIdx.x) * 4u;   // covers DM*DM elems
  f32x4 a = *(const f32x4*)(W0 + i);
  f32x4 b = *(const f32x4*)(W1 + i);
  u32x2 pa, pb;
  pa.x = f2bf(a.x) | (f2bf(a.y) << 16);
  pa.y = f2bf(a.z) | (f2bf(a.w) << 16);
  pb.x = f2bf(b.x) | (f2bf(b.y) << 16);
  pb.y = f2bf(b.z) | (f2bf(b.w) << 16);
  *(u32x2*)(Wb0 + i) = pa;
  *(u32x2*)(Wb1 + i) = pb;
}

// ---- gathered dual-expert GEMM --------------------------------------------
// grid = (M/32, 2), block = 256 (8 waves).  Each block: 32 tokens (2 M-tiles),
// N range = blockIdx.y*512 .. +511.  Each wave: 4 N-tiles x 2 M-tiles x 2 roles.
// B fragments are loaded once per (k,t) and reused by both M-tiles.
// All global addressing uses uniform base + 32-bit lane offsets (GVS mode).
__global__ __launch_bounds__(256) void role_gemm_kernel(
    const int* __restrict__ ids, const int* __restrict__ role,
    const float* __restrict__ emb,
    const unsigned short* __restrict__ Wb0,
    const unsigned short* __restrict__ Wb1,
    float* __restrict__ out)
{
  __shared__ __align__(16) unsigned short Abf[MROWS * LDS_STRIDE];

  const int tid   = threadIdx.x;
  const int mBase = blockIdx.x * MROWS;

  // ---- gather 32 embedding rows, convert to bf16 into LDS ----
  {
    const int r  = tid >> 3;            // row 0..31
    const int c0 = (tid & 7) * 128;     // 128 floats per thread
    const unsigned row = (unsigned)ids[mBase + r];
    const unsigned eOff = (row << 10) + (unsigned)c0;   // 32-bit elem offset (max ~51.5M)
    unsigned short* dst = &Abf[r * LDS_STRIDE + c0];
#pragma unroll
    for (int i = 0; i < 32; ++i) {
      f32x4 f = *(const f32x4*)(emb + eOff + i * 4);
      u32x2 p;
      p.x = f2bf(f.x) | (f2bf(f.y) << 16);
      p.y = f2bf(f.z) | (f2bf(f.w) << 16);
      *(u32x2*)(dst + i * 4) = p;
    }
  }
  __syncthreads();

  const int lane  = tid & 31;
  const int wave  = tid >> 5;                 // 0..7
  const int hw    = lane >> 4;                // half-wave: 0 or 1
  const int l15   = lane & 15;
  const int nBase = blockIdx.y * 512 + wave * 64;

  v8f acc0[2][4] = {};      // [mtile][ntile], expert 0
  v8f acc1[2][4] = {};      // expert 1

  const unsigned short* aRow0 = &Abf[l15 * LDS_STRIDE];
  const unsigned short* aRow1 = &Abf[(16 + l15) * LDS_STRIDE];

  // per-lane 32-bit element offsets into Wb rows (column-major B fragment)
  unsigned bOff[4];
#pragma unroll
  for (int t = 0; t < 4; ++t)
    bOff[t] = ((unsigned)(nBase + t * 16 + l15) << 10) + (unsigned)(8 * hw);

  for (int k0 = 0; k0 < DM; k0 += KSTEP) {
    // A fragments (16x32 bf16): lane holds M=l15, K chunks {k0+8*hw..+7}, {k0+16+8*hw..+7}
    FragU a0, a1;
    a0.q[0] = *(const u32x4*)(aRow0 + k0 + 8 * hw);
    a0.q[1] = *(const u32x4*)(aRow0 + k0 + 16 + 8 * hw);
    a1.q[0] = *(const u32x4*)(aRow1 + k0 + 8 * hw);
    a1.q[1] = *(const u32x4*)(aRow1 + k0 + 16 + 8 * hw);

#pragma unroll
    for (int t = 0; t < 4; ++t) {
      const unsigned o = bOff[t] + (unsigned)k0;
      // B fragment (32x16 bf16): lane holds column N, K chunks {k0+8*hw..+7}, {k0+16+8*hw..+7}
      FragU b0, b1;
      b0.q[0] = *(const u32x4*)(Wb0 + o);
      b0.q[1] = *(const u32x4*)(Wb0 + o + 16);
      b1.q[0] = *(const u32x4*)(Wb1 + o);
      b1.q[1] = *(const u32x4*)(Wb1 + o + 16);

      acc0[0][t] = __builtin_amdgcn_wmma_f32_16x16x32_bf16(
          false, a0.v, false, b0.v, (short)0, acc0[0][t], false, false);
      acc1[0][t] = __builtin_amdgcn_wmma_f32_16x16x32_bf16(
          false, a0.v, false, b1.v, (short)0, acc1[0][t], false, false);
      acc0[1][t] = __builtin_amdgcn_wmma_f32_16x16x32_bf16(
          false, a1.v, false, b0.v, (short)0, acc0[1][t], false, false);
      acc1[1][t] = __builtin_amdgcn_wmma_f32_16x16x32_bf16(
          false, a1.v, false, b1.v, (short)0, acc1[1][t], false, false);
    }
  }

  // ---- select by role and store (non-temporal: streamed once) ----
#pragma unroll
  for (int mt = 0; mt < 2; ++mt) {
    int rflag[8];
#pragma unroll
    for (int r = 0; r < 8; ++r)
      rflag[r] = role[mBase + mt * 16 + r + 8 * hw];
#pragma unroll
    for (int t = 0; t < 4; ++t) {
      const unsigned n = (unsigned)(nBase + t * 16 + l15);
#pragma unroll
      for (int r = 0; r < 8; ++r) {
        const unsigned m = (unsigned)(mBase + mt * 16 + r + 8 * hw);
        const float v = rflag[r] ? acc1[mt][t][r] : acc0[mt][t][r];
        __builtin_nontemporal_store(v, out + (m << 10) + n);
      }
    }
  }
}

// ---- launch ----------------------------------------------------------------
extern "C" void kernel_launch(void* const* d_in, const int* in_sizes, int n_in,
                              void* d_out, int out_size, void* d_ws, size_t ws_size,
                              hipStream_t stream) {
  (void)n_in; (void)out_size; (void)ws_size;
  const int*   ids  = (const int*)d_in[0];
  const int*   role = (const int*)d_in[1];
  const float* emb  = (const float*)d_in[2];
  const float* W0   = (const float*)d_in[3];
  const float* W1   = (const float*)d_in[4];
  float*       out  = (float*)d_out;

  unsigned short* Wb0 = (unsigned short*)d_ws;
  unsigned short* Wb1 = Wb0 + (size_t)DM * DM;

  // convert weights: DM*DM elems, 4 per thread, 256 threads/block
  cvt_w_kernel<<<(DM * DM) / (256 * 4), 256, 0, stream>>>(W0, W1, Wb0, Wb1);

  const int nTokens = in_sizes[0];          // B*L = 16384
  dim3 grid(nTokens / MROWS, 2);
  role_gemm_kernel<<<grid, 256, 0, stream>>>(ids, role, emb, Wb0, Wb1, out);
}
